// CausalAttention_27058293964911
// MI455X (gfx1250) — matrix operand order repaired and checked by reading, not compiled, not run on hardware
//
#include <hip/hip_runtime.h>
#include <hip/hip_bf16.h>
#include <math.h>

// ---------------------------------------------------------------------------
// CausalAttention for MI455X (gfx1250): bf16 WMMA pipeline.
//   qkv = x @ w_qkv^T            (WMMA bf16 -> f32)
//   q,k : l2norm + rope          (f32 math, bf16 out)
//   flash causal attention       (WMMA bf16, online softmax in LDS)
//   out = o @ w_o^T              (WMMA bf16 -> f32)
// ---------------------------------------------------------------------------

typedef __bf16 bf16;
typedef __attribute__((ext_vector_type(8)))  __bf16 bf16x8;
typedef __attribute__((ext_vector_type(16))) __bf16 bf16x16;
typedef __attribute__((ext_vector_type(8)))  float  f32x8;

#define B_SZ 4
#define T_SEQ 2048
#define D_MODEL 2048
#define N_HEADS 16
#define N_KV 4
#define HEAD_DIM 128
#define QKV_OUT 3072
#define SOFTMAX_SCALE 0.08838834764831845f
#define QK_EPS 1e-6f

// 16 contiguous bf16 split as two 16B loads (works for global and LDS ptrs).
__device__ __forceinline__ bf16x16 load_frag(const bf16* p0, const bf16* p1) {
    bf16x16 r;
    *reinterpret_cast<bf16x8*>(&r)       = *reinterpret_cast<const bf16x8*>(p0);
    *(reinterpret_cast<bf16x8*>(&r) + 1) = *reinterpret_cast<const bf16x8*>(p1);
    return r;
}

__device__ __forceinline__ f32x8 wmma_bf16(bf16x16 a, bf16x16 b, f32x8 c) {
    // D = A(16x32) * B(32x16) + C, f32 accumulate
    return __builtin_amdgcn_wmma_f32_16x16x32_bf16(
        false, a, false, b, (short)0, c, false, false);
}

// ---------------------------------------------------------------------------
// f32 -> bf16 elementwise cast
// ---------------------------------------------------------------------------
__global__ void cast_f32_to_bf16(const float* __restrict__ in,
                                 bf16* __restrict__ out, int n) {
    int i = blockIdx.x * blockDim.x + threadIdx.x;
    if (i < n) out[i] = (bf16)in[i];
}

// ---------------------------------------------------------------------------
// GEMM C(MxN, f32) = A(MxK, bf16 row-major) * B(NxK, bf16 row-major)^T
// WG = 256 threads = 8 waves laid out 4(M) x 2(N); wave tile = 32x32.
// ---------------------------------------------------------------------------
__global__ __launch_bounds__(256) void gemm_bf16_nt(
    const bf16* __restrict__ A, const bf16* __restrict__ B,
    float* __restrict__ C, int M, int N, int K)
{
    const int lane = threadIdx.x & 31;
    const int wave = threadIdx.x >> 5;
    const int l16  = lane & 15;
    const int half = lane >> 4;       // which 16-lane half
    const int aoff = half * 16;       // A-frag K base per lane half
    const int wm = wave & 3, wn = wave >> 2;
    const int row0 = blockIdx.y * 128 + wm * 32;
    const int col0 = blockIdx.x * 64  + wn * 32;

    f32x8 c00 = {}, c01 = {}, c10 = {}, c11 = {};

    const bf16* a0p = A + (size_t)(row0 + l16) * K;
    const bf16* a1p = A + (size_t)(row0 + 16 + l16) * K;
    const bf16* b0p = B + (size_t)(col0 + l16) * K;
    const bf16* b1p = B + (size_t)(col0 + 16 + l16) * K;

    for (int k0 = 0; k0 < K; k0 += 32) {
        bf16x16 a0 = load_frag(a0p + k0 + aoff, a0p + k0 + aoff + 8);
        bf16x16 a1 = load_frag(a1p + k0 + aoff, a1p + k0 + aoff + 8);
        bf16x16 b0 = load_frag(b0p + k0 + half * 16, b0p + k0 + half * 16 + 8);
        bf16x16 b1 = load_frag(b1p + k0 + half * 16, b1p + k0 + half * 16 + 8);
        c00 = wmma_bf16(a0, b0, c00);
        c01 = wmma_bf16(a0, b1, c01);
        c10 = wmma_bf16(a1, b0, c10);
        c11 = wmma_bf16(a1, b1, c11);
    }

    // C/D layout: lane half gives rows r+8*half, column = l16 (+ tile offset)
    #pragma unroll
    for (int r = 0; r < 8; ++r) {
        C[(size_t)(row0 + r + half * 8) * N + col0 + l16]           = c00[r];
        C[(size_t)(row0 + r + half * 8) * N + col0 + 16 + l16]      = c01[r];
        C[(size_t)(row0 + 16 + r + half * 8) * N + col0 + l16]      = c10[r];
        C[(size_t)(row0 + 16 + r + half * 8) * N + col0 + 16 + l16] = c11[r];
    }
}

// ---------------------------------------------------------------------------
// Q/K: l2-normalize over head_dim then RoPE; write bf16 (b,h,t,d).
// One thread per (b, t, head) over 20 heads (16 Q + 4 K).
// ---------------------------------------------------------------------------
__global__ void qk_norm_rope(const float* __restrict__ qkv,
                             bf16* __restrict__ Qb, bf16* __restrict__ Kb)
{
    int g = blockIdx.x * blockDim.x + threadIdx.x;
    const int total = B_SZ * T_SEQ * 20;
    if (g >= total) return;
    int b   = g / (T_SEQ * 20);
    int rem = g % (T_SEQ * 20);
    int t   = rem / 20;
    int hh  = rem % 20;

    const float* src;
    bf16* dst;
    if (hh < N_HEADS) {
        src = qkv + ((size_t)(b * T_SEQ + t)) * QKV_OUT + hh * HEAD_DIM;
        dst = Qb + ((size_t)((b * N_HEADS + hh) * T_SEQ + t)) * HEAD_DIM;
    } else {
        int kvh = hh - N_HEADS;
        src = qkv + ((size_t)(b * T_SEQ + t)) * QKV_OUT + D_MODEL + kvh * HEAD_DIM;
        dst = Kb + ((size_t)((b * N_KV + kvh) * T_SEQ + t)) * HEAD_DIM;
    }

    float ss = 0.f;
    for (int d = 0; d < HEAD_DIM; ++d) { float x = src[d]; ss += x * x; }
    float inv = 1.0f / fmaxf(sqrtf(ss), QK_EPS);

    const float half_dim = (float)(HEAD_DIM / 2);
    for (int d = 0; d < HEAD_DIM / 2; ++d) {
        float invf = powf(10000.0f, -(float)d / half_dim);
        float ang  = (float)t * invf;
        float c = cosf(ang), s = sinf(ang);
        float x1 = src[d] * inv;
        float x2 = src[d + HEAD_DIM / 2] * inv;
        dst[d]                = (bf16)(x1 * c - x2 * s);
        dst[d + HEAD_DIM / 2] = (bf16)(x1 * s + x2 * c);
    }
}

// ---------------------------------------------------------------------------
// V: transpose (b,t,kv,d) -> bf16 (b,kv,d,t) so P*V B-fragments load
// 16 contiguous K (=t) values per lane.
// ---------------------------------------------------------------------------
__global__ void v_transpose(const float* __restrict__ qkv, bf16* __restrict__ Vt)
{
    int t  = blockIdx.x * blockDim.x + threadIdx.x;  // 0..2047
    int d  = blockIdx.y;                             // 0..127
    int bk = blockIdx.z;                             // b*N_KV + kv
    int b  = bk >> 2, kvh = bk & 3;
    float v = qkv[((size_t)(b * T_SEQ + t)) * QKV_OUT + D_MODEL + N_KV * HEAD_DIM
                  + kvh * HEAD_DIM + d];
    Vt[((size_t)bk * HEAD_DIM + d) * T_SEQ + t] = (bf16)v;
}

// ---------------------------------------------------------------------------
// Flash causal attention. Grid (T/128, H, B); 8 waves/WG, one 16-row Q tile
// per wave. Online softmax over 32-wide K blocks; scores go through per-wave
// LDS for the C-frag -> A-frag layout swap. Output bf16 (b,t, h*128+d).
// ---------------------------------------------------------------------------
__global__ __launch_bounds__(256) void flash_attn(
    const bf16* __restrict__ Qb, const bf16* __restrict__ Kb,
    const bf16* __restrict__ Vt, bf16* __restrict__ Ob)
{
    __shared__ float sS[8][16][32];   // raw scores
    __shared__ bf16  sP[8][16][32];   // probabilities (bf16 for A-frag reload)
    __shared__ float sStat[8][16];    // per-row corr / 1/sum broadcast

    const int lane = threadIdx.x & 31;
    const int wave = threadIdx.x >> 5;
    const int l16  = lane & 15;
    const int half = lane >> 4;
    const int aoff = half * 16;

    const int b  = blockIdx.z;
    const int h  = blockIdx.y;
    const int kv = h >> 2;                       // jnp.repeat(k, 4, axis=2)
    const int q0 = blockIdx.x * 128 + wave * 16;

    const bf16* qbase = Qb + ((size_t)(b * N_HEADS + h) * T_SEQ + q0 + l16) * HEAD_DIM;
    const bf16* kbase = Kb + ((size_t)(b * N_KV + kv) * T_SEQ) * HEAD_DIM;
    const bf16* vbase = Vt + ((size_t)(b * N_KV + kv) * HEAD_DIM) * T_SEQ;

    // Preload Q tile: 4 A-fragments covering head_dim=128 (K=32 each).
    bf16x16 qf[4];
    #pragma unroll
    for (int dc = 0; dc < 4; ++dc)
        qf[dc] = load_frag(qbase + dc * 32 + aoff, qbase + dc * 32 + aoff + 8);

    f32x8 o[8];
    #pragma unroll
    for (int f = 0; f < 8; ++f) o[f] = {};

    float rowMax = -__builtin_inff();   // valid in lanes 0..15 (row = l16)
    float rowSum = 0.f;

    const int nblk = (q0 + 47) >> 5;    // K blocks of 32 covering k <= q0+15
    for (int kb = 0; kb < nblk; ++kb) {
        const int k0 = kb * 32;

        // ---- S = Q * K^T : two 16x16 C-frags (k cols 0-15 / 16-31) ----
        f32x8 slo = {}, shi = {};
        #pragma unroll
        for (int dc = 0; dc < 4; ++dc) {
            const bf16* kp0 = kbase + (size_t)(k0 + l16) * HEAD_DIM + dc * 32 + half * 16;
            bf16x16 bl = load_frag(kp0, kp0 + 8);
            slo = wmma_bf16(qf[dc], bl, slo);
            const bf16* kp1 = kbase + (size_t)(k0 + 16 + l16) * HEAD_DIM + dc * 32 + half * 16;
            bf16x16 bh = load_frag(kp1, kp1 + 8);
            shi = wmma_bf16(qf[dc], bh, shi);
        }
        #pragma unroll
        for (int r = 0; r < 8; ++r) {
            sS[wave][r + half * 8][l16]      = slo[r];
            sS[wave][r + half * 8][16 + l16] = shi[r];
        }
        asm volatile("s_wait_dscnt 0x0" ::: "memory");

        // ---- row-wise online softmax (lanes 0..15, row = l16) ----
        if (lane < 16) {
            const int q = q0 + l16;
            float bm = -__builtin_inff();
            float sc[32];
            #pragma unroll
            for (int j = 0; j < 32; ++j) {
                float s = sS[wave][l16][j] * SOFTMAX_SCALE;
                s = (k0 + j <= q) ? s : -__builtin_inff();
                sc[j] = s;
                bm = fmaxf(bm, s);
            }
            float newM = fmaxf(rowMax, bm);          // finite: block has >=1 valid
            float corr = __expf(rowMax - newM);      // exp(-inf)=0 on first block
            float psum = 0.f;
            #pragma unroll
            for (int j = 0; j < 32; ++j) {
                float p = __expf(sc[j] - newM);
                sP[wave][l16][j] = (bf16)p;
                psum += p;
            }
            rowSum = rowSum * corr + psum;
            rowMax = newM;
            sStat[wave][l16] = corr;
        }
        asm volatile("s_wait_dscnt 0x0" ::: "memory");

        // ---- rescale accumulators by per-row corr ----
        float cr[8];
        #pragma unroll
        for (int r = 0; r < 8; ++r) cr[r] = sStat[wave][r + half * 8];
        #pragma unroll
        for (int f = 0; f < 8; ++f)
            #pragma unroll
            for (int r = 0; r < 8; ++r) o[f][r] *= cr[r];

        // ---- O += P * V : P as A-frag from LDS, V^T as B-frags ----
        bf16x16 pf = load_frag(&sP[wave][l16][aoff], &sP[wave][l16][aoff + 8]);
        #pragma unroll
        for (int f = 0; f < 8; ++f) {
            const bf16* vp = vbase + (size_t)(f * 16 + l16) * T_SEQ + k0 + half * 16;
            bf16x16 vf = load_frag(vp, vp + 8);
            o[f] = wmma_bf16(pf, vf, o[f]);
        }
    }

    // ---- finalize: divide by row sums, emit bf16 at (b, q, h*128+d) ----
    if (lane < 16) sStat[wave][l16] = 1.0f / rowSum;
    asm volatile("s_wait_dscnt 0x0" ::: "memory");
    float invs[8];
    #pragma unroll
    for (int r = 0; r < 8; ++r) invs[r] = sStat[wave][r + half * 8];

    bf16* obase = Ob + (size_t)b * T_SEQ * D_MODEL;
    #pragma unroll
    for (int f = 0; f < 8; ++f)
        #pragma unroll
        for (int r = 0; r < 8; ++r)
            obase[(size_t)(q0 + r + half * 8) * D_MODEL + h * HEAD_DIM + f * 16 + l16] =
                (bf16)(o[f][r] * invs[r]);
}

// ---------------------------------------------------------------------------
// Host-side orchestration (all launches on `stream`, graph-capture safe).
// ---------------------------------------------------------------------------
extern "C" void kernel_launch(void* const* d_in, const int* in_sizes, int n_in,
                              void* d_out, int out_size, void* d_ws, size_t ws_size,
                              hipStream_t stream) {
    const float* x     = (const float*)d_in[0];   // (4, 2048, 2048)
    const float* w_qkv = (const float*)d_in[1];   // (3072, 2048)
    const float* w_o   = (const float*)d_in[2];   // (2048, 2048)
    float* out = (float*)d_out;                   // (4, 2048, 2048) f32
    char* ws = (char*)d_ws;

    const size_t N_X   = (size_t)B_SZ * T_SEQ * D_MODEL;      // 16.8M
    const size_t N_WQ  = (size_t)QKV_OUT * D_MODEL;           // 6.3M
    const size_t N_WO  = (size_t)D_MODEL * D_MODEL;           // 4.2M
    const size_t N_QKV = (size_t)B_SZ * T_SEQ * QKV_OUT;      // 25.2M

    size_t off = 0;
    bf16*  xb   = (bf16*)(ws + off); off += N_X * 2;          //  32 MB
    bf16*  wqb  = (bf16*)(ws + off); off += N_WQ * 2;         //  12 MB
    bf16*  wob  = (bf16*)(ws + off); off += N_WO * 2;         //   8 MB
    float* qkv  = (float*)(ws + off); off += N_QKV * 4;       //  96 MB
    bf16*  Qb   = (bf16*)(ws + off); off += (size_t)B_SZ * N_HEADS * T_SEQ * HEAD_DIM * 2;
    bf16*  Kb   = (bf16*)(ws + off); off += (size_t)B_SZ * N_KV * T_SEQ * HEAD_DIM * 2;
    bf16*  Vt   = (bf16*)(ws + off); off += (size_t)B_SZ * N_KV * HEAD_DIM * T_SEQ * 2;
    bf16*  Ob   = (bf16*)qkv;  // alias: qkv dead after prep kernels

    // 1) casts to bf16
    cast_f32_to_bf16<<<(int)((N_X  + 255) / 256), 256, 0, stream>>>(x, xb, (int)N_X);
    cast_f32_to_bf16<<<(int)((N_WQ + 255) / 256), 256, 0, stream>>>(w_qkv, wqb, (int)N_WQ);
    cast_f32_to_bf16<<<(int)((N_WO + 255) / 256), 256, 0, stream>>>(w_o, wob, (int)N_WO);

    // 2) qkv = x @ w_qkv^T   (M=8192, N=3072, K=2048)
    gemm_bf16_nt<<<dim3(QKV_OUT / 64, (B_SZ * T_SEQ) / 128), 256, 0, stream>>>(
        xb, wqb, qkv, B_SZ * T_SEQ, QKV_OUT, D_MODEL);

    // 3) q/k norm+rope, v transpose
    qk_norm_rope<<<(B_SZ * T_SEQ * 20 + 127) / 128, 128, 0, stream>>>(qkv, Qb, Kb);
    v_transpose<<<dim3(T_SEQ / 256, HEAD_DIM, B_SZ * N_KV), 256, 0, stream>>>(qkv, Vt);

    // 4) flash causal attention -> Ob (b, t, h*128+d) bf16
    flash_attn<<<dim3(T_SEQ / 128, N_HEADS, B_SZ), 256, 0, stream>>>(Qb, Kb, Vt, Ob);

    // 5) out = Ob @ w_o^T   (M=8192, N=2048, K=2048) -> f32 d_out
    gemm_bf16_nt<<<dim3(D_MODEL / 64, (B_SZ * T_SEQ) / 128), 256, 0, stream>>>(
        Ob, wob, out, B_SZ * T_SEQ, D_MODEL, D_MODEL);
}
